// MambaBlock_41051297415686
// MI455X (gfx1250) — matrix operand order repaired and checked by reading, not compiled, not run on hardware
//
#include <hip/hip_runtime.h>
#include <hip/hip_bf16.h>

// ---------------------------------------------------------------------------
// Mamba block for gfx1250 (MI455X): bf16 WMMA GEMMs with async LDS staging
// (GLOBAL_LOAD_ASYNC_TO_LDS_B128 + s_wait_asynccnt) + fp32 scan.
// ---------------------------------------------------------------------------

typedef __bf16 bf16_t;
typedef __bf16 v16bf __attribute__((ext_vector_type(16)));
typedef float  v8f   __attribute__((ext_vector_type(8)));

#define H_DIM   1024
#define INNER_D 2048
#define NSTATE  16
#define RDIM    64
#define BATCH   2
#define LSEQ    4096
#define MROWS   (BATCH * LSEQ)   // 8192

__device__ __forceinline__ bf16_t f2bf(float f) {
  union { float f; unsigned u; } v; v.f = f;
  unsigned r = v.u + 0x7FFFu + ((v.u >> 16) & 1u);   // round-to-nearest-even
  unsigned short h = (unsigned short)(r >> 16);
  bf16_t out;
  __builtin_memcpy(&out, &h, 2);
  return out;
}

__device__ __forceinline__ float siluf(float v) {
  return v / (1.0f + __expf(-v));
}

__device__ __forceinline__ float softplusf(float v) {
  return v > 20.0f ? v : log1pf(__expf(v));
}

// ---- CDNA5 async global->LDS copy (ASYNCcnt path) -------------------------
__device__ __forceinline__ void async_copy_b128(const void* gsrc, void* ldst) {
#if __has_builtin(__builtin_amdgcn_global_load_async_to_lds_b128)
  typedef int v4i_t __attribute__((vector_size(16)));
  __builtin_amdgcn_global_load_async_to_lds_b128(
      (__attribute__((address_space(1))) v4i_t*)gsrc,
      (__attribute__((address_space(3))) v4i_t*)ldst,
      0, 0);
#else
  unsigned lds_off = (unsigned)(unsigned long long)ldst;  // low 32b = LDS offset
  asm volatile("global_load_async_to_lds_b128 %0, %1, off"
               :: "v"(lds_off), "v"(gsrc) : "memory");
#endif
}

__device__ __forceinline__ void wait_async0() {
#if __has_builtin(__builtin_amdgcn_s_wait_asynccnt)
  __builtin_amdgcn_s_wait_asynccnt(0);
#else
  asm volatile("s_wait_asynccnt 0x0" ::: "memory");
#endif
}

// ---------------------------------------------------------------------------
// Generic fp32 -> bf16 conversion (weights)
// ---------------------------------------------------------------------------
__global__ void cvt_bf16_kernel(const float* __restrict__ src,
                                bf16_t* __restrict__ dst, int n) {
  int i = blockIdx.x * blockDim.x + threadIdx.x;
  if (i < n) dst[i] = f2bf(src[i]);
}

// ---------------------------------------------------------------------------
// LayerNorm over H=1024, write bf16
// ---------------------------------------------------------------------------
__global__ __launch_bounds__(256)
void layernorm_bf16_kernel(const float* __restrict__ x,
                           const float* __restrict__ w,
                           const float* __restrict__ b,
                           bf16_t* __restrict__ out) {
  __shared__ float red[256];
  const int row = blockIdx.x;
  const float* xr = x + (size_t)row * H_DIM;
  float s = 0.f, s2 = 0.f;
  for (int i = threadIdx.x; i < H_DIM; i += 256) {
    float v = xr[i];
    s += v; s2 += v * v;
  }
  red[threadIdx.x] = s; __syncthreads();
  for (int o = 128; o > 0; o >>= 1) {
    if (threadIdx.x < o) red[threadIdx.x] += red[threadIdx.x + o];
    __syncthreads();
  }
  const float mean = red[0] * (1.0f / H_DIM);
  __syncthreads();
  red[threadIdx.x] = s2; __syncthreads();
  for (int o = 128; o > 0; o >>= 1) {
    if (threadIdx.x < o) red[threadIdx.x] += red[threadIdx.x + o];
    __syncthreads();
  }
  const float var  = red[0] * (1.0f / H_DIM) - mean * mean;
  const float rstd = rsqrtf(var + 1e-5f);
  for (int i = threadIdx.x; i < H_DIM; i += 256) {
    float v = (xr[i] - mean) * rstd * w[i] + b[i];
    out[(size_t)row * H_DIM + i] = f2bf(v);
  }
}

// ---------------------------------------------------------------------------
// bf16 WMMA GEMM: out[M x N] = A[M x K] * W[N x K]^T  (+ epilogue)
//   EPI 0: + bias (nullable)
//   EPI 2: softplus(acc + bias)
//   EPI 3: acc + bias + resid  (final out_proj + residual)
// Block tile 128x128, KT=64, double-buffered async LDS staging.
// 8 waves (wave32) -> each wave 64x32 via 4x2 WMMA tiles x 2 k-steps.
// ---------------------------------------------------------------------------
template <int EPI>
__global__ __launch_bounds__(256)
void gemm_bf16_wmma_kernel(const bf16_t* __restrict__ A,
                           const bf16_t* __restrict__ W,
                           const float*  __restrict__ bias,
                           const float*  __restrict__ resid,
                           float* __restrict__ out,
                           int M, int N, int K) {
  constexpr int KT  = 64;
  constexpr int LDS = 72;  // 64 + 8 pad elems -> 144 B rows (16B aligned)
  __shared__ __align__(16) bf16_t sA[2][128 * LDS];
  __shared__ __align__(16) bf16_t sB[2][128 * LDS];

  const int tid  = threadIdx.x;
  const int lane = tid & 31;
  const int wave = tid >> 5;      // 0..7
  const int wm   = wave >> 2;     // 0..1 : 64-row strip
  const int wn   = wave & 3;      // 0..3 : 32-col strip
  const int half = lane >> 4;     // 0/1
  const int l16  = lane & 15;

  const int m0 = blockIdx.y * 128;
  const int n0 = blockIdx.x * 128;

  v8f acc[4][2];
  #pragma unroll
  for (int i = 0; i < 4; ++i)
    #pragma unroll
    for (int j = 0; j < 2; ++j)
      #pragma unroll
      for (int e = 0; e < 8; ++e) acc[i][j][e] = 0.0f;

  // cooperative stage mapping: each thread fills 32 elems (64 B) of A and B
  const int srow = tid & 127;              // tile row (A) / out-col row (B)
  const int sseg = (tid >> 7) & 1;         // which 32-elem segment of 64
  const int nclamp = (n0 + srow < N) ? (n0 + srow) : (N - 1);

  const bf16_t* gA = A + (size_t)(m0 + srow) * K + sseg * 32;
  const bf16_t* gW = W + (size_t)nclamp   * K + sseg * 32;
  bf16_t* lA0 = &sA[0][srow * LDS + sseg * 32];
  bf16_t* lB0 = &sB[0][srow * LDS + sseg * 32];
  bf16_t* lA1 = &sA[1][srow * LDS + sseg * 32];
  bf16_t* lB1 = &sB[1][srow * LDS + sseg * 32];

  auto stage = [&](int buf, int k0) {
    bf16_t* la = buf ? lA1 : lA0;
    bf16_t* lb = buf ? lB1 : lB0;
    #pragma unroll
    for (int j = 0; j < 4; ++j) {   // 4 x b128 = 64 B
      async_copy_b128(gA + k0 + j * 8, la + j * 8);
      async_copy_b128(gW + k0 + j * 8, lb + j * 8);
    }
  };

  stage(0, 0);
  wait_async0();
  __syncthreads();

  for (int k0 = 0; k0 < K; k0 += KT) {
    const int cur = (k0 >> 6) & 1;
    if (k0 + KT < K) stage(cur ^ 1, k0 + KT);   // async prefetch overlaps WMMA

    const bf16_t* bA = sA[cur];
    const bf16_t* bB = sB[cur];
    #pragma unroll
    for (int ks = 0; ks < 2; ++ks) {            // two 32-wide k-steps
      const int kb = ks * 32;
      // B fragments: lane<16 -> col=l16, k[0,16); lane>=16 -> k[16,32)
      v16bf bfrag[2];
      #pragma unroll
      for (int j = 0; j < 2; ++j) {
        const bf16_t* p = bB + (wn * 32 + j * 16 + l16) * LDS + kb + half * 16;
        union { uint4 u[2]; v16bf v; } ub;
        ub.u[0] = *(const uint4*)p;
        ub.u[1] = *(const uint4*)(p + 8);
        bfrag[j] = ub.v;
      }
      // A fragments + WMMA
      #pragma unroll
      for (int i = 0; i < 4; ++i) {
        const bf16_t* p = bA + (wm * 64 + i * 16 + l16) * LDS + kb + half * 16;
        union { uint4 u[2]; v16bf v; } ua;
        ua.u[0] = *(const uint4*)p;        // k = half*16 + [0,8)
        ua.u[1] = *(const uint4*)(p + 8);  // k = half*16 + 8 + [0,8)
        v16bf afrag = ua.v;
        #pragma unroll
        for (int j = 0; j < 2; ++j)
          acc[i][j] = __builtin_amdgcn_wmma_f32_16x16x32_bf16(
              false, afrag, false, bfrag[j], (short)0, acc[i][j], false, false);
      }
    }
    wait_async0();       // this wave's prefetch done
    __syncthreads();     // all waves' prefetches done / tile reads finished
  }

  // ---- epilogue: C layout VGPR r -> (M = r + 8*half, N = l16) ----
  #pragma unroll
  for (int i = 0; i < 4; ++i) {
    const int rowb = m0 + wm * 64 + i * 16 + half * 8;
    #pragma unroll
    for (int j = 0; j < 2; ++j) {
      const int col = n0 + wn * 32 + j * 16 + l16;
      if (col < N) {
        const float bv = bias ? bias[col] : 0.0f;
        #pragma unroll
        for (int r = 0; r < 8; ++r) {
          const int rr = rowb + r;
          float v = acc[i][j][r] + bv;
          if (EPI == 2) v = softplusf(v);
          if (EPI == 3) v += resid[(size_t)rr * N + col];
          out[(size_t)rr * N + col] = v;
        }
      }
    }
  }
}

// ---------------------------------------------------------------------------
// Depthwise causal conv (K=4) + SiLU; reads x-half of proj, writes fp32 + bf16
// ---------------------------------------------------------------------------
__global__ __launch_bounds__(256)
void conv_silu_kernel(const float* __restrict__ proj,
                      const float* __restrict__ cw,
                      const float* __restrict__ cb,
                      float* __restrict__ xf,
                      bf16_t* __restrict__ xb) {
  const size_t idx = (size_t)blockIdx.x * blockDim.x + threadIdx.x;
  if (idx >= (size_t)MROWS * INNER_D) return;
  const int d   = (int)(idx % INNER_D);
  const size_t ml = idx / INNER_D;          // m = b*L + l
  const int l   = (int)(ml % LSEQ);
  float acc = cb[d];
  #pragma unroll
  for (int j = 0; j < 4; ++j) {
    const int ls = l - 3 + j;
    if (ls >= 0)
      acc += cw[d * 4 + j] * proj[(ml + (size_t)(ls - l)) * (2 * INNER_D) + d];
  }
  const float sv = siluf(acc);
  xf[idx] = sv;
  xb[idx] = f2bf(sv);
}

// ---------------------------------------------------------------------------
// Extract dt_raw (first 64 cols of params) to bf16 for the dt GEMM
// ---------------------------------------------------------------------------
__global__ void extract_dtraw_kernel(const float* __restrict__ params,
                                     bf16_t* __restrict__ dtb) {
  const int idx = blockIdx.x * blockDim.x + threadIdx.x;
  if (idx >= MROWS * RDIM) return;
  const int m = idx >> 6, k = idx & 63;
  dtb[idx] = f2bf(params[m * (RDIM + 2 * NSTATE) + k]);
}

// ---------------------------------------------------------------------------
// Selective scan: thread per (b, d); n=0..15 in registers; B/C staged in LDS.
// Fuses gate SiLU multiply; writes y in bf16 for the out_proj GEMM.
// ---------------------------------------------------------------------------
#define TCH 64
__global__ __launch_bounds__(256)
void scan_kernel(const float* __restrict__ dt,
                 const float* __restrict__ x,
                 const float* __restrict__ params,
                 const float* __restrict__ proj,
                 const float* __restrict__ A_log,
                 const float* __restrict__ Dp,
                 bf16_t* __restrict__ yb) {
  __shared__ float sBC[TCH][32];
  const int b = blockIdx.y;
  const int d = blockIdx.x * 256 + threadIdx.x;

  float a[NSTATE], st[NSTATE];
  #pragma unroll
  for (int n = 0; n < NSTATE; ++n) {
    a[n]  = -__expf(A_log[d * NSTATE + n]);
    st[n] = 0.0f;
  }
  const float Dv = Dp[d];
  const size_t base = (size_t)b * LSEQ;

  for (int t0 = 0; t0 < LSEQ; t0 += TCH) {
    for (int i = threadIdx.x; i < TCH * 32; i += 256) {
      const int tt = i >> 5, c = i & 31;
      sBC[tt][c] = params[(base + t0 + tt) * (RDIM + 2 * NSTATE) + RDIM + c];
    }
    __syncthreads();
    for (int i = 0; i < TCH; ++i) {
      const size_t m = base + t0 + i;
      const float dtv = dt[m * INNER_D + d];
      const float xv  = x[m * INNER_D + d];
      const float dx  = dtv * xv;
      float y = Dv * xv;
      #pragma unroll
      for (int n = 0; n < NSTATE; ++n) {
        const float dec = __expf(dtv * a[n]);
        st[n] = dec * st[n] + dx * sBC[i][n];
        y += st[n] * sBC[i][16 + n];
      }
      const float g = proj[m * (2 * INNER_D) + INNER_D + d];
      yb[m * INNER_D + d] = f2bf(y * siluf(g));
    }
    __syncthreads();
  }
}

// ---------------------------------------------------------------------------
// Launch
// ---------------------------------------------------------------------------
extern "C" void kernel_launch(void* const* d_in, const int* in_sizes, int n_in,
                              void* d_out, int out_size, void* d_ws, size_t ws_size,
                              hipStream_t stream) {
  (void)in_sizes; (void)n_in; (void)out_size; (void)ws_size;

  const float* hidden = (const float*)d_in[0];
  const float* ln_w   = (const float*)d_in[1];
  const float* ln_b   = (const float*)d_in[2];
  const float* in_w   = (const float*)d_in[3];
  const float* in_b   = (const float*)d_in[4];
  const float* conv_w = (const float*)d_in[5];
  const float* conv_b = (const float*)d_in[6];
  const float* xp_w   = (const float*)d_in[7];
  const float* dtp_w  = (const float*)d_in[8];
  const float* dtp_b  = (const float*)d_in[9];
  const float* A_log  = (const float*)d_in[10];
  const float* Dp     = (const float*)d_in[11];
  const float* out_w  = (const float*)d_in[12];
  const float* out_b  = (const float*)d_in[13];

  char* ws = (char*)d_ws;
  size_t off = 0;
  auto alloc = [&](size_t bytes) -> void* {
    void* p = ws + off;
    off = (off + bytes + 255) & ~(size_t)255;
    return p;
  };

  bf16_t* hn_bf    = (bf16_t*)alloc((size_t)MROWS * H_DIM * 2);
  bf16_t* inw_bf   = (bf16_t*)alloc((size_t)(2 * INNER_D) * H_DIM * 2);
  bf16_t* xpw_bf   = (bf16_t*)alloc((size_t)(RDIM + 2 * NSTATE) * INNER_D * 2);
  bf16_t* dtpw_bf  = (bf16_t*)alloc((size_t)INNER_D * RDIM * 2);
  bf16_t* outw_bf  = (bf16_t*)alloc((size_t)H_DIM * INNER_D * 2);
  float*  proj     = (float*) alloc((size_t)MROWS * (2 * INNER_D) * 4);
  float*  xf       = (float*) alloc((size_t)MROWS * INNER_D * 4);
  bf16_t* xb       = (bf16_t*)alloc((size_t)MROWS * INNER_D * 2);
  float*  params   = (float*) alloc((size_t)MROWS * (RDIM + 2 * NSTATE) * 4);
  bf16_t* dtraw_bf = (bf16_t*)alloc((size_t)MROWS * RDIM * 2);
  float*  dtf      = (float*) alloc((size_t)MROWS * INNER_D * 4);
  bf16_t* yb       = (bf16_t*)alloc((size_t)MROWS * INNER_D * 2);

  // Weight conversions (fp32 -> bf16)
  {
    int n;
    n = 2 * INNER_D * H_DIM;
    cvt_bf16_kernel<<<(n + 255) / 256, 256, 0, stream>>>(in_w, inw_bf, n);
    n = (RDIM + 2 * NSTATE) * INNER_D;
    cvt_bf16_kernel<<<(n + 255) / 256, 256, 0, stream>>>(xp_w, xpw_bf, n);
    n = INNER_D * RDIM;
    cvt_bf16_kernel<<<(n + 255) / 256, 256, 0, stream>>>(dtp_w, dtpw_bf, n);
    n = H_DIM * INNER_D;
    cvt_bf16_kernel<<<(n + 255) / 256, 256, 0, stream>>>(out_w, outw_bf, n);
  }

  // 1) LayerNorm -> bf16
  layernorm_bf16_kernel<<<MROWS, 256, 0, stream>>>(hidden, ln_w, ln_b, hn_bf);

  // 2) in_proj: proj = hn @ in_w^T + in_b   [8192 x 4096, K=1024]
  gemm_bf16_wmma_kernel<0><<<dim3((2 * INNER_D) / 128, MROWS / 128), 256, 0, stream>>>(
      hn_bf, inw_bf, in_b, nullptr, proj, MROWS, 2 * INNER_D, H_DIM);

  // 3) depthwise conv + SiLU
  conv_silu_kernel<<<(unsigned)(((size_t)MROWS * INNER_D) / 256), 256, 0, stream>>>(
      proj, conv_w, conv_b, xf, xb);

  // 4) x-proj: params = x @ xp_w^T   [8192 x 96, K=2048]
  gemm_bf16_wmma_kernel<0><<<dim3(1, MROWS / 128), 256, 0, stream>>>(
      xb, xpw_bf, nullptr, nullptr, params, MROWS, RDIM + 2 * NSTATE, INNER_D);

  // 5) dt = softplus(dt_raw @ dtp_w^T + dtp_b)   [8192 x 2048, K=64]
  extract_dtraw_kernel<<<(MROWS * RDIM) / 256, 256, 0, stream>>>(params, dtraw_bf);
  gemm_bf16_wmma_kernel<2><<<dim3(INNER_D / 128, MROWS / 128), 256, 0, stream>>>(
      dtraw_bf, dtpw_bf, dtp_b, nullptr, dtf, MROWS, INNER_D, RDIM);

  // 6) selective scan + gate
  scan_kernel<<<dim3(INNER_D / 256, BATCH), 256, 0, stream>>>(
      dtf, xf, params, proj, A_log, Dp, yb);

  // 7) out_proj + residual: out = hidden + y @ out_w^T + out_b  [8192 x 1024, K=2048]
  gemm_bf16_wmma_kernel<3><<<dim3(H_DIM / 128, MROWS / 128), 256, 0, stream>>>(
      yb, outw_bf, out_b, hidden, (float*)d_out, MROWS, H_DIM, INNER_D);
}
